// FlashMultiHeadAttention_59631325938236
// MI455X (gfx1250) — matrix operand order repaired and checked by reading, not compiled, not run on hardware
//
#include <hip/hip_runtime.h>
#include <hip/hip_bf16.h>

typedef __attribute__((ext_vector_type(16))) __bf16 v16bf;
typedef __attribute__((ext_vector_type(8)))  __bf16 v8bf;
typedef __attribute__((ext_vector_type(2)))  __bf16 v2bf;
typedef __attribute__((ext_vector_type(8)))  float  v8f;

#define WMMA_BF16(a, b, c) \
  __builtin_amdgcn_wmma_f32_16x16x32_bf16(false, (a), false, (b), (short)0, (c), false, false)

static constexpr int Bb = 2, Ss = 2048, Dd = 1024, Hh = 16, HDd = 64;

// ---------------- fragment loaders: all-contiguous 128-bit LDS reads --------

// A-matrix 16x32 bf16 from row-major tile [m][k] (ld multiple of 8 elems).
__device__ __forceinline__ v16bf ld_frag_a(const __bf16* tile, int ld) {
  const int lane = threadIdx.x & 31;
  const int row  = lane & 15;
  const int kofs = (lane & 16) ? 8 : 0;
  const v8bf lo = *(const v8bf*)(tile + row * ld + kofs);
  const v8bf hi = *(const v8bf*)(tile + row * ld + kofs + 16);
  return __builtin_shufflevector(lo, hi, 0, 1, 2, 3, 4, 5, 6, 7,
                                 8, 9, 10, 11, 12, 13, 14, 15);
}

// B-matrix 32x16 bf16 from a TRANSPOSED tile: tileT[n][k].
__device__ __forceinline__ v16bf ld_frag_b(const __bf16* tileT, int ld) {
  const int lane = threadIdx.x & 31;
  const int col  = lane & 15;
  const int kofs = (lane & 16) ? 8 : 0;
  const v8bf lo = *(const v8bf*)(tileT + col * ld + kofs);
  const v8bf hi = *(const v8bf*)(tileT + col * ld + kofs + 16);
  return __builtin_shufflevector(lo, hi, 0, 1, 2, 3, 4, 5, 6, 7,
                                 8, 9, 10, 11, 12, 13, 14, 15);
}

// ---------------- one-time prep kernels -------------------------------------

__global__ void __launch_bounds__(256) cvt_f32_bf16(const float* __restrict__ in,
                                                    __bf16* __restrict__ out, int n) {
  int i = (blockIdx.x * 256 + threadIdx.x) * 4;
  if (i >= n) return;
  float4 f = *(const float4*)(in + i);
  out[i + 0] = (__bf16)f.x;
  out[i + 1] = (__bf16)f.y;
  out[i + 2] = (__bf16)f.z;
  out[i + 3] = (__bf16)f.w;
}

// in[R][C] f32 -> out[C][R] bf16 (R, C multiples of 32)
__global__ void __launch_bounds__(256) transpose_cvt(const float* __restrict__ in,
                                                     __bf16* __restrict__ out,
                                                     int R, int C) {
  __shared__ __bf16 t[32][33];
  const int c0 = blockIdx.x * 32, r0 = blockIdx.y * 32;
  const int tx = threadIdx.x & 31, ty = threadIdx.x >> 5;  // ty: 0..7
#pragma unroll
  for (int i = 0; i < 32; i += 8)
    t[ty + i][tx] = (__bf16)in[(size_t)(r0 + ty + i) * C + c0 + tx];
  __syncthreads();
#pragma unroll
  for (int i = 0; i < 32; i += 8)
    out[(size_t)(c0 + ty + i) * R + r0 + tx] = t[tx][ty + i];
}

// ---------------- all-bf16 GEMM:  C[M,N] = A[M,K] * Bt[N,K]^T + bias --------
// Block tile 64(M) x 128(N) x 32(K); 8 waves, each computes 32x32 (4 wmma/step).
// Double-buffered LDS: stage tile it+1 while computing tile it; 1 barrier/step.
template <bool QKV_SCATTER>
__global__ void __launch_bounds__(256) gemm_bf16(
    const __bf16* __restrict__ Ap, const __bf16* __restrict__ Bt,
    const float* __restrict__ bias, float* __restrict__ Cout,
    __bf16* __restrict__ qw, __bf16* __restrict__ kw, __bf16* __restrict__ vw,
    int M, int N, int K) {
  __shared__ alignas(16) __bf16 sA [2][64][40];    // [m][k]
  __shared__ alignas(16) __bf16 sBt[2][128][40];   // [n][k]

  const int tid  = threadIdx.x;
  const int lane = tid & 31;
  const int wid  = tid >> 5;
  const int wm   = wid & 1;   // 2 wave-rows of 32
  const int wn   = wid >> 1;  // 4 wave-cols of 32
  const int m0   = blockIdx.y * 64;
  const int n0   = blockIdx.x * 128;

  const int ra_st = tid >> 2, ca_st = (tid & 3) * 8;  // A staging coords

  auto stage = [&](int buf, int k0) {
    *(v8bf*)(&sA[buf][ra_st][ca_st]) =
        *(const v8bf*)(Ap + (size_t)(m0 + ra_st) * K + k0 + ca_st);
#pragma unroll
    for (int i = 0; i < 2; ++i) {
      int lin = tid + 256 * i;
      int r = lin >> 2, ck = (lin & 3) * 8;
      *(v8bf*)(&sBt[buf][r][ck]) =
          *(const v8bf*)(Bt + (size_t)(n0 + r) * K + k0 + ck);
    }
  };

  v8f c00 = {}, c01 = {}, c10 = {}, c11 = {};

  stage(0, 0);
  __syncthreads();

  const int nk = K >> 5;
  for (int it = 0; it < nk; ++it) {
    if (it + 1 < nk) stage((it + 1) & 1, (it + 1) << 5);  // prefetch next tile

    const int cur = it & 1;
    v16bf a0 = ld_frag_a(&sA[cur][wm * 32][0], 40);
    v16bf a1 = ld_frag_a(&sA[cur][wm * 32 + 16][0], 40);
    v16bf b0 = ld_frag_b(&sBt[cur][wn * 32][0], 40);
    v16bf b1 = ld_frag_b(&sBt[cur][wn * 32 + 16][0], 40);
    c00 = WMMA_BF16(a0, b0, c00);
    c01 = WMMA_BF16(a0, b1, c01);
    c10 = WMMA_BF16(a1, b0, c10);
    c11 = WMMA_BF16(a1, b1, c11);
    __syncthreads();  // next tile staged AND current tile reads done
  }

  auto emit = [&](int gm, int gn, float vacc) {
    float v = vacc + bias[gn];
    if constexpr (QKV_SCATTER) {
      int bi = gm >> 11, s = gm & (Ss - 1);
      int w = gn >> 10, cc = gn & 1023;
      int h = cc >> 6, hd = cc & 63;
      if (w == 0)
        qw[(((size_t)bi * Hh + h) * Ss + s) * HDd + hd] = (__bf16)v;
      else if (w == 1)
        kw[(((size_t)bi * Hh + h) * Ss + s) * HDd + hd] = (__bf16)v;
      else  // V stored transposed per head: [b,h,hd,s]
        vw[(((size_t)bi * Hh + h) * HDd + hd) * Ss + s] = (__bf16)v;
    } else {
      Cout[(size_t)gm * N + gn] = v;
    }
  };

#pragma unroll
  for (int e = 0; e < 8; ++e) {
    int c  = lane & 15;
    int r  = e + ((lane & 16) ? 8 : 0);
    int ra = m0 + wm * 32 + r, rb = ra + 16;
    int ca = n0 + wn * 32 + c, cb = ca + 16;
    emit(ra, ca, c00[e]);
    emit(ra, cb, c01[e]);
    emit(rb, ca, c10[e]);
    emit(rb, cb, c11[e]);
  }
}

// ---------------- flash attention: one WG per (b*H+h, 64-row Q tile) --------
// Q,K layout [b,h,s,64]; V layout [b,h,64,s] (pre-transposed by QKV epilogue).
__global__ void __launch_bounds__(256) flash_attn(
    const __bf16* __restrict__ Q, const __bf16* __restrict__ Kt,
    const __bf16* __restrict__ Vt, __bf16* __restrict__ O) {
  __shared__ alignas(16) __bf16 sQ [64][72];   // [s][hd]     A for QK^T
  __shared__ alignas(16) __bf16 sK [64][72];   // [kvpos][hd] = [n][k] for QK^T
  __shared__ alignas(16) __bf16 sVt[64][72];   // [hd][kvpos] = [n][k] for P@V
  __shared__ alignas(16) __bf16 sP [64][72];   // [s][kvpos]  A for P@V
  __shared__ float  sS[64][68];
  __shared__ float  sM[64], sL[64], sAl[64];

  const int tid  = threadIdx.x;
  const int lane = tid & 31;
  const int wid  = tid >> 5;
  const int wm   = wid & 3;
  const int wn   = wid >> 2;
  const int bh   = blockIdx.y;       // b*H + h
  const int q0   = blockIdx.x * 64;

  const size_t head = (size_t)bh * Ss * HDd;
  const __bf16* Qb  = Q  + head + (size_t)q0 * HDd;
  const __bf16* VbT = Vt + head;     // [64][Ss]

  if (tid < 64) { sM[tid] = -1e30f; sL[tid] = 0.f; }

#pragma unroll
  for (int i = 0; i < 2; ++i) {      // Q tile 64x64: 16B chunks
    int lin = tid + 256 * i;
    int r = lin >> 3, ck = (lin & 7) * 8;
    *(v8bf*)(&sQ[r][ck]) = *(const v8bf*)(Qb + r * 64 + ck);
  }

  v8f o0 = {}, o1 = {};
  const float scale = 0.125f;        // 1/sqrt(64)

  for (int j = 0; j < Ss; j += 64) {
    const __bf16* Kb = Kt + head + (size_t)j * HDd;
#pragma unroll
    for (int i = 0; i < 2; ++i) {
      int lin = tid + 256 * i;
      int r = lin >> 3, ck = (lin & 7) * 8;
      *(v8bf*)(&sK[r][ck])  = *(const v8bf*)(Kb + r * 64 + ck);
      *(v8bf*)(&sVt[r][ck]) = *(const v8bf*)(VbT + (size_t)r * Ss + j + ck);
    }
    __syncthreads();

    // S = Q @ K^T
    v8f s0 = {}, s1 = {};
#pragma unroll
    for (int kk = 0; kk < 64; kk += 32) {
      v16bf a  = ld_frag_a(&sQ[wm * 16][kk], 72);
      v16bf b0 = ld_frag_b(&sK[wn * 32][kk], 72);
      v16bf b1 = ld_frag_b(&sK[wn * 32 + 16][kk], 72);
      s0 = WMMA_BF16(a, b0, s0);
      s1 = WMMA_BF16(a, b1, s1);
    }
#pragma unroll
    for (int e = 0; e < 8; ++e) {
      int c = lane & 15;
      int r = e + ((lane & 16) ? 8 : 0);
      sS[wm * 16 + r][wn * 32 + c]      = s0[e] * scale;
      sS[wm * 16 + r][wn * 32 + 16 + c] = s1[e] * scale;
    }
    __syncthreads();

    // online softmax: one thread per row; packed bf16 P stores
    if (tid < 64) {
      float mold = sM[tid];
      float mx = mold;
#pragma unroll
      for (int c = 0; c < 64; ++c) mx = fmaxf(mx, sS[tid][c]);
      float alpha = __expf(mold - mx);
      float sum = 0.f;
#pragma unroll
      for (int c = 0; c < 64; c += 2) {
        float p0 = __expf(sS[tid][c] - mx);
        float p1 = __expf(sS[tid][c + 1] - mx);
        v2bf pk;
        pk[0] = (__bf16)p0;
        pk[1] = (__bf16)p1;
        *(v2bf*)(&sP[tid][c]) = pk;
        sum += p0 + p1;
      }
      sL[tid]  = sL[tid] * alpha + sum;
      sM[tid]  = mx;
      sAl[tid] = alpha;
    }
    __syncthreads();

    // rescale running O, then O += P @ V
#pragma unroll
    for (int e = 0; e < 8; ++e) {
      int r = e + ((lane & 16) ? 8 : 0);
      float al = sAl[wm * 16 + r];
      o0[e] *= al;
      o1[e] *= al;
    }
#pragma unroll
    for (int kk = 0; kk < 64; kk += 32) {
      v16bf a  = ld_frag_a(&sP[wm * 16][kk], 72);
      v16bf b0 = ld_frag_b(&sVt[wn * 32][kk], 72);
      v16bf b1 = ld_frag_b(&sVt[wn * 32 + 16][kk], 72);
      o0 = WMMA_BF16(a, b0, o0);
      o1 = WMMA_BF16(a, b1, o1);
    }
    __syncthreads();
  }

  // normalize and store to [b, s, h*64 + col] bf16 workspace
  const int b = bh >> 4, h = bh & 15;
#pragma unroll
  for (int e = 0; e < 8; ++e) {
    int c = lane & 15;
    int r = e + ((lane & 16) ? 8 : 0);
    int grow  = q0 + wm * 16 + r;
    float inv = 1.f / sL[wm * 16 + r];
    size_t base = ((size_t)b * Ss + grow) * (Hh * HDd) + h * HDd + wn * 32 + c;
    O[base]      = (__bf16)(o0[e] * inv);
    O[base + 16] = (__bf16)(o1[e] * inv);
  }
}

// ---------------------------------------------------------------------------
extern "C" void kernel_launch(void* const* d_in, const int* in_sizes, int n_in,
                              void* d_out, int out_size, void* d_ws, size_t ws_size,
                              hipStream_t stream) {
  (void)in_sizes; (void)n_in; (void)out_size; (void)ws_size;
  const float* x     = (const float*)d_in[0];
  const float* w_qkv = (const float*)d_in[1];
  const float* b_qkv = (const float*)d_in[2];
  const float* w_out = (const float*)d_in[3];
  const float* b_out = (const float*)d_in[4];
  float* out = (float*)d_out;

  const size_t MB = 1u << 20;
  char* ws = (char*)d_ws;
  __bf16* qws  = (__bf16*)(ws + 0 * MB);    // [b,h,s,64]      8 MB
  __bf16* kws  = (__bf16*)(ws + 8 * MB);    // [b,h,s,64]      8 MB
  __bf16* vws  = (__bf16*)(ws + 16 * MB);   // [b,h,64,s]      8 MB
  __bf16* aws  = (__bf16*)(ws + 24 * MB);   // [B*S,1024]      8 MB
  __bf16* xb   = (__bf16*)(ws + 32 * MB);   // [4096,1024]     8 MB
  __bf16* wqT  = (__bf16*)(ws + 40 * MB);   // [3072,1024]     6 MB
  __bf16* woT  = (__bf16*)(ws + 46 * MB);   // [1024,1024]     2 MB

  dim3 blk(256);
  // one-time converts / transposes
  cvt_f32_bf16<<<dim3(Bb * Ss * Dd / 1024), blk, 0, stream>>>(x, xb, Bb * Ss * Dd);
  transpose_cvt<<<dim3(3072 / 32, 1024 / 32), blk, 0, stream>>>(w_qkv, wqT, 1024, 3072);
  transpose_cvt<<<dim3(1024 / 32, 1024 / 32), blk, 0, stream>>>(w_out, woT, 1024, 1024);

  // 1) QKV projection -> scatter q/k [b,h,s,64], v transposed [b,h,64,s]
  gemm_bf16<true><<<dim3(3072 / 128, 4096 / 64), blk, 0, stream>>>(
      xb, wqT, b_qkv, nullptr, qws, kws, vws, Bb * Ss, 3 * Hh * HDd, Dd);
  // 2) flash attention per head
  flash_attn<<<dim3(Ss / 64, Bb * Hh), blk, 0, stream>>>(qws, kws, vws, aws);
  // 3) output projection + bias -> f32 out
  gemm_bf16<false><<<dim3(1024 / 128, 4096 / 64), blk, 0, stream>>>(
      aws, woT, b_out, out, nullptr, nullptr, nullptr, Bb * Ss, Hh * HDd, Hh * HDd);
}
